// Head_87497073754273
// MI455X (gfx1250) — matrix operand order repaired and checked by reading, not compiled, not run on hardware
//
#include <hip/hip_runtime.h>
#include <hip/hip_bf16.h>

// ---------------------------------------------------------------------------
// Single-head attention for MI455X (gfx1250).
//   Phase 0: convert Wq/Wk/Wv fp32 -> bf16 (once, L2-resident).
//   Phase 1: bf16 WMMA projections; vh stored transposed (vhT[b][v][s]).
//   Phase 2: flash attention; K/V tiles staged in LDS by the Tensor Data
//            Mover (tensor_load_to_lds, TENSORcnt), double-buffered.
// ---------------------------------------------------------------------------

typedef __attribute__((ext_vector_type(16))) __bf16 v16bf;
typedef __attribute__((ext_vector_type(8)))  __bf16 v8bf;
typedef __attribute__((ext_vector_type(8)))  float  v8f;
typedef __attribute__((ext_vector_type(4)))  unsigned int su4;
typedef __attribute__((ext_vector_type(8)))  unsigned int su8;

#define DIN    1024
#define DK     128
#define SEQ    4096
#define NBAT   8
#define AWAVES 8   // waves per block in attention kernel

__device__ __forceinline__ __bf16 f2bf(float f) { return (__bf16)f; }
__device__ __forceinline__ unsigned rfl(unsigned x) {
  return (unsigned)__builtin_amdgcn_readfirstlane((int)x);
}

__device__ __forceinline__ v8f wmma_bf16(v16bf a, v16bf b, v8f c) {
  return __builtin_amdgcn_wmma_f32_16x16x32_bf16(false, a, false, b,
                                                 (short)0, c, false, false);
}

// --------------------------- TDM descriptor helpers ------------------------
// D# group 0: count=1 | lds_addr | global_addr | type=2 (ISA 08 §8.3)
__device__ __forceinline__ su4 tdm_g0(unsigned lds_addr, unsigned long long ga) {
  su4 g;
  g.x = 1u;                                   // count=1, no gather
  g.y = rfl(lds_addr);                        // LDS byte address
  g.z = rfl((unsigned)ga);                    // global_addr[31:0]
  g.w = rfl((unsigned)(ga >> 32) | (2u << 30)); // global_addr[56:32] | type=2
  return g;
}
// D# group 1: data_size=1 (2B), dims/strides in elements (ISA 08 §8.4)
__device__ __forceinline__ su8 tdm_g1(unsigned td0, unsigned td1,
                                      unsigned tl0, unsigned tl1,
                                      unsigned long long st0,
                                      unsigned long long st1) {
  su8 g;
  g[0] = 1u << 16;                                  // wg_mask=0, data_size=2B
  g[1] = (td0 & 0xFFFFu) << 16;                     // tensor_dim0[15:0]
  g[2] = (td0 >> 16) | ((td1 & 0xFFFFu) << 16);     // td0[31:16] | td1[15:0]
  g[3] = (td1 >> 16) | (tl0 << 16);                 // td1[31:16] | tile_dim0
  g[4] = (tl1 & 0xFFFFu);                           // tile_dim1, tile_dim2=0
  g[5] = (unsigned)st0;                             // dim0_stride[31:0]
  g[6] = ((unsigned)(st0 >> 32) & 0xFFFFu) |        // dim0_stride[47:32]
         (((unsigned)st1 & 0xFFFFu) << 16);         // dim1_stride[15:0]
  g[7] = (unsigned)(st1 >> 16);                     // dim1_stride[47:16]
  return g;
}
__device__ __forceinline__ void tdm_load(su4 g0, su8 g1) {
  asm volatile("tensor_load_to_lds %0, %1" :: "s"(g0), "s"(g1) : "memory");
}

// ---------------------------------------------------------------------------
// Phase 0: W fp32 -> bf16 (3 x 128 x 1024).
// ---------------------------------------------------------------------------
__global__ __launch_bounds__(256) void wcvt_kernel(
    const float* __restrict__ Wq, const float* __restrict__ Wk,
    const float* __restrict__ Wv, __bf16* __restrict__ WB)
{
  const int sel = blockIdx.y;
  const float* __restrict__ W = (sel == 0) ? Wq : (sel == 1) ? Wk : Wv;
  __bf16* __restrict__ dst = WB + (size_t)sel * DK * DIN;
  const int i = (blockIdx.x * 256 + threadIdx.x) * 8;   // 64 blocks x 256 thr
  const float4 f0 = *reinterpret_cast<const float4*>(W + i);
  const float4 f1 = *reinterpret_cast<const float4*>(W + i + 4);
  v8bf o;
  o[0]=f2bf(f0.x); o[1]=f2bf(f0.y); o[2]=f2bf(f0.z); o[3]=f2bf(f0.w);
  o[4]=f2bf(f1.x); o[5]=f2bf(f1.y); o[6]=f2bf(f1.z); o[7]=f2bf(f1.w);
  *reinterpret_cast<v8bf*>(dst + i) = o;
}

// ---------------------------------------------------------------------------
// Phase 1: y = x @ W^T (fp32 x, bf16 W). Wave: 16 rows x 128 cols.
// ---------------------------------------------------------------------------
__global__ __launch_bounds__(256) void proj_qkv_kernel(
    const float* __restrict__ Xq, const float* __restrict__ Xk,
    const float* __restrict__ Xv, const __bf16* __restrict__ WB,
    __bf16* __restrict__ QH, __bf16* __restrict__ KH, __bf16* __restrict__ VT)
{
  const int lane = threadIdx.x & 31;
  const int wave = threadIdx.x >> 5;
  const int ln   = lane & 15;
  const int hf   = lane >> 4;
  const int sel  = blockIdx.y;  // 0=q, 1=k, 2=v

  const float*  __restrict__ X  = (sel == 0) ? Xq : (sel == 1) ? Xk : Xv;
  const __bf16* __restrict__ Wb = WB + (size_t)sel * DK * DIN;

  const int row0 = blockIdx.x * 128 + wave * 16;
  const size_t mrow = (size_t)(row0 + ln) * DIN;

  v8f acc[8];
  #pragma unroll
  for (int t = 0; t < 8; ++t) acc[t] = (v8f){0.f,0.f,0.f,0.f,0.f,0.f,0.f,0.f};

  for (int ks = 0; ks < DIN / 32; ++ks) {
    const int K0 = ks * 32;
    // A fragment (convert x fp32 -> bf16 on the fly)
    v16bf a;
    {
      const float* ap = X + mrow + K0 + hf * 16;
      const float4 f0 = *reinterpret_cast<const float4*>(ap);
      const float4 f1 = *reinterpret_cast<const float4*>(ap + 4);
      const float4 f2 = *reinterpret_cast<const float4*>(ap + 8);
      const float4 f3 = *reinterpret_cast<const float4*>(ap + 12);
      a[0]=f2bf(f0.x); a[1]=f2bf(f0.y); a[2]=f2bf(f0.z); a[3]=f2bf(f0.w);
      a[4]=f2bf(f1.x); a[5]=f2bf(f1.y); a[6]=f2bf(f1.z); a[7]=f2bf(f1.w);
      a[8]=f2bf(f2.x); a[9]=f2bf(f2.y); a[10]=f2bf(f2.z); a[11]=f2bf(f2.w);
      a[12]=f2bf(f3.x); a[13]=f2bf(f3.y); a[14]=f2bf(f3.z); a[15]=f2bf(f3.w);
    }
    // B fragments straight from bf16 W (lane = n, halves e -> K = hf*16 + e)
    #pragma unroll
    for (int t = 0; t < 8; ++t) {
      const v16bf bfr = *reinterpret_cast<const v16bf*>(
          Wb + (size_t)(t * 16 + ln) * DIN + K0 + hf * 16);
      acc[t] = wmma_bf16(a, bfr, acc[t]);
    }
  }

  if (sel < 2) {
    __bf16* __restrict__ Y = (sel == 0) ? QH : KH;
    #pragma unroll
    for (int t = 0; t < 8; ++t)
      #pragma unroll
      for (int r = 0; r < 8; ++r)
        Y[(size_t)(row0 + r + 8 * hf) * DK + t * 16 + ln] = f2bf(acc[t][r]);
  } else {
    const int bb = row0 / SEQ;
    const int s0 = row0 % SEQ;
    #pragma unroll
    for (int t = 0; t < 8; ++t) {
      v8bf vb;
      #pragma unroll
      for (int r = 0; r < 8; ++r) vb[r] = f2bf(acc[t][r]);
      __bf16* dst = VT + ((size_t)bb * DK + t * 16 + ln) * SEQ + s0 + 8 * hf;
      *reinterpret_cast<v8bf*>(dst) = vb;
    }
  }
}

// ---------------------------------------------------------------------------
// Phase 2: flash attention; TDM stages K (32x128, contiguous 8KB) and
// V (128 rows x 32 cols of vhT, strided) into double-buffered LDS.
// ---------------------------------------------------------------------------
__global__ __launch_bounds__(AWAVES * 32) void flash_attn_kernel(
    const __bf16* __restrict__ QH, const __bf16* __restrict__ KH,
    const __bf16* __restrict__ VT, float* __restrict__ OUT)
{
  __shared__ __align__(32) __bf16 kbuf[2][32][DK];
  __shared__ __align__(32) __bf16 vbuf[2][DK][32];
  __shared__ __align__(32) __bf16 pbuf[AWAVES][16][32];

  const int lane = threadIdx.x & 31;
  const int wave = threadIdx.x >> 5;
  const int ln   = lane & 15;
  const int hf   = lane >> 4;
  const int bb   = blockIdx.y;
  const int q0   = blockIdx.x * (AWAVES * 16) + wave * 16;
  const size_t bRow = (size_t)bb * SEQ;

  // scalar (SGPR) leader test so non-leader waves branch over the TDM ops
  const bool leader = (__builtin_amdgcn_readfirstlane((int)threadIdx.x) < 32);

  const unsigned kld[2] = { (unsigned)(uintptr_t)&kbuf[0][0][0],
                            (unsigned)(uintptr_t)&kbuf[1][0][0] };
  const unsigned vld[2] = { (unsigned)(uintptr_t)&vbuf[0][0][0],
                            (unsigned)(uintptr_t)&vbuf[1][0][0] };
  const unsigned long long kga0 =
      (unsigned long long)(uintptr_t)(KH + bRow * DK);
  const unsigned long long vga0 =
      (unsigned long long)(uintptr_t)(VT + (size_t)bb * DK * SEQ);
  // K: 1-D copy of 4096 elems (8KB). V: 128 rows x 32 elems, row stride SEQ.
  const su8 g1k = tdm_g1(1u << 22, 1u, 32u * DK, 0u, 1ull << 22, 0ull);
  const su8 g1v = tdm_g1(SEQ, DK, 32u, DK, (unsigned long long)SEQ, 0ull);

  // Q fragments: 16 rows x 128, kept in VGPRs.
  v16bf aq[4];
  {
    const __bf16* qp = QH + (bRow + q0 + ln) * DK;
    #pragma unroll
    for (int kk = 0; kk < 4; ++kk) {
      v8bf lo = *reinterpret_cast<const v8bf*>(qp + kk * 32 + hf * 16);
      v8bf hi = *reinterpret_cast<const v8bf*>(qp + kk * 32 + hf * 16 + 8);
      #pragma unroll
      for (int j = 0; j < 8; ++j) { aq[kk][j] = lo[j]; aq[kk][8 + j] = hi[j]; }
    }
  }

  v8f Oacc[8];
  #pragma unroll
  for (int t = 0; t < 8; ++t) Oacc[t] = (v8f){0.f,0.f,0.f,0.f,0.f,0.f,0.f,0.f};
  float mstate[8], lstate[8];
  #pragma unroll
  for (int r = 0; r < 8; ++r) { mstate[r] = -__builtin_inff(); lstate[r] = 0.f; }

  const float scale = 0.08838834764831845f;  // 1/sqrt(128)
  const int nIter = SEQ / 32;

  if (leader) {  // prologue: tile 0 into buffer 0
    tdm_load(tdm_g0(kld[0], kga0), g1k);
    tdm_load(tdm_g0(vld[0], vga0), g1v);
  }

  for (int it = 0; it < nIter; ++it) {
    const int cur = it & 1;
    if (leader) {
      if (it + 1 < nIter) {  // prefetch next tile into the other buffer
        tdm_load(tdm_g0(kld[cur ^ 1], kga0 + (size_t)(it + 1) * 32 * DK * 2), g1k);
        tdm_load(tdm_g0(vld[cur ^ 1], vga0 + (size_t)(it + 1) * 32 * 2), g1v);
        __builtin_amdgcn_s_wait_tensorcnt(2);  // current tile's pair done
      } else {
        __builtin_amdgcn_s_wait_tensorcnt(0);
      }
    }
    __syncthreads();  // publish kbuf[cur]/vbuf[cur] to all waves

    // ---- scores: 16 x 32 (two 16x16 C tiles), K fragments from LDS ----
    v8f sc[2];
    #pragma unroll
    for (int j = 0; j < 2; ++j) {
      sc[j] = (v8f){0.f,0.f,0.f,0.f,0.f,0.f,0.f,0.f};
      const __bf16* kp = &kbuf[cur][j * 16 + ln][0];
      #pragma unroll
      for (int kk = 0; kk < 4; ++kk) {
        v16bf bk = *reinterpret_cast<const v16bf*>(kp + kk * 32 + hf * 16);
        sc[j] = wmma_bf16(aq[kk], bk, sc[j]);
      }
    }

    // ---- online softmax ----
    float corr[8];
    #pragma unroll
    for (int r = 0; r < 8; ++r) {
      float a0 = sc[0][r] * scale;
      float a1 = sc[1][r] * scale;
      float mx = fmaxf(a0, a1);
      mx = fmaxf(mx, __shfl_xor(mx, 1, 32));
      mx = fmaxf(mx, __shfl_xor(mx, 2, 32));
      mx = fmaxf(mx, __shfl_xor(mx, 4, 32));
      mx = fmaxf(mx, __shfl_xor(mx, 8, 32));
      float mnew = fmaxf(mstate[r], mx);
      float c = __expf(mstate[r] - mnew);
      corr[r] = c;
      mstate[r] = mnew;
      float p0 = __expf(a0 - mnew);
      float p1 = __expf(a1 - mnew);
      sc[0][r] = p0;
      sc[1][r] = p1;
      float ts = p0 + p1;
      ts += __shfl_xor(ts, 1, 32);
      ts += __shfl_xor(ts, 2, 32);
      ts += __shfl_xor(ts, 4, 32);
      ts += __shfl_xor(ts, 8, 32);
      lstate[r] = lstate[r] * c + ts;
    }
    #pragma unroll
    for (int t = 0; t < 8; ++t)
      #pragma unroll
      for (int r = 0; r < 8; ++r) Oacc[t][r] *= corr[r];

    // ---- P (C layout) -> A fragment via per-wave LDS bounce ----
    #pragma unroll
    for (int j = 0; j < 2; ++j)
      #pragma unroll
      for (int r = 0; r < 8; ++r)
        pbuf[wave][r + 8 * hf][j * 16 + ln] = f2bf(sc[j][r]);
    __builtin_amdgcn_wave_barrier();
    v16bf pa;
    {
      const __bf16* pp = &pbuf[wave][ln][0];
      v8bf lo = *reinterpret_cast<const v8bf*>(pp + hf * 16);
      v8bf hi = *reinterpret_cast<const v8bf*>(pp + hf * 16 + 8);
      #pragma unroll
      for (int j = 0; j < 8; ++j) { pa[j] = lo[j]; pa[8 + j] = hi[j]; }
    }
    __builtin_amdgcn_wave_barrier();

    // ---- O += P @ V, V fragments from LDS (vbuf[v][kv]) ----
    #pragma unroll
    for (int t = 0; t < 8; ++t) {
      v16bf bv = *reinterpret_cast<const v16bf*>(&vbuf[cur][t * 16 + ln][hf * 16]);
      Oacc[t] = wmma_bf16(pa, bv, Oacc[t]);
    }

    __syncthreads();  // all waves done with buffers before TDM overwrites
  }

  // ---- normalize and store fp32 output ----
  #pragma unroll
  for (int r = 0; r < 8; ++r) {
    const float inv = 1.0f / lstate[r];
    const size_t orow = (bRow + q0 + r + 8 * hf) * DK;
    #pragma unroll
    for (int t = 0; t < 8; ++t)
      OUT[orow + t * 16 + ln] = Oacc[t][r] * inv;
  }
}

// ---------------------------------------------------------------------------
extern "C" void kernel_launch(void* const* d_in, const int* in_sizes, int n_in,
                              void* d_out, int out_size, void* d_ws, size_t ws_size,
                              hipStream_t stream) {
  const float* q  = (const float*)d_in[0];
  const float* k  = (const float*)d_in[1];
  const float* v  = (const float*)d_in[2];
  const float* Wq = (const float*)d_in[3];
  const float* Wk = (const float*)d_in[4];
  const float* Wv = (const float*)d_in[5];

  const size_t nProj = (size_t)NBAT * SEQ * DK;
  __bf16* QH = (__bf16*)d_ws;
  __bf16* KH = QH + nProj;
  __bf16* VT = KH + nProj;
  __bf16* WB = VT + nProj;  // 3 * 128 * 1024 bf16

  dim3 g0(64, 3), b0(256);
  wcvt_kernel<<<g0, b0, 0, stream>>>(Wq, Wk, Wv, WB);

  dim3 g1(NBAT * SEQ / 128, 3), b1(256);
  proj_qkv_kernel<<<g1, b1, 0, stream>>>(q, k, v, WB, QH, KH, VT);

  dim3 g2(SEQ / (AWAVES * 16), NBAT), b2(AWAVES * 32);
  flash_attn_kernel<<<g2, b2, 0, stream>>>(QH, KH, VT, (float*)d_out);
}
